// S4Wrapper_25039659335759
// MI455X (gfx1250) — matrix operand order repaired and checked by reading, not compiled, not run on hardware
//
#include <hip/hip_runtime.h>
#include <math.h>

// Problem constants (match reference)
#define Bsz 128
#define Hd  1024
#define Ls  256
#define Ns  64

typedef _Float16 f16;
typedef __attribute__((ext_vector_type(16))) _Float16 v16h;
typedef __attribute__((ext_vector_type(8)))  float    v8f;

__device__ __forceinline__ float frcp(float x) {
  return __builtin_amdgcn_rcpf(x);              // v_rcp_f32, branch-free
}
__device__ __forceinline__ float gelu_fast(float x) {
  // tanh-GELU == x * sigmoid(2*u): one v_exp_f32 + one v_rcp_f32
  float u = 0.7978845608028654f * (x + 0.044715f * x * x * x);
  return x * frcp(1.0f + __expf(-2.0f * u));
}
__device__ __forceinline__ float sigmoid_fast(float x) {
  return frcp(1.0f + __expf(-x));
}

// ---------------------------------------------------------------------------
// Kernel 1: build S4D convolution kernel k(h, l) with ZOH discretization.
// k[h,l] = 2 * Re( sum_n C_disc[h,n] * exp(dtA[h,n] * l) )
// One block per channel h; thread t computes position l = t.
// Uses hardware transcendentals (v_exp/v_sin/v_cos) throughout.
// ---------------------------------------------------------------------------
__global__ __launch_bounds__(256) void s4d_kernel_build(
    const float* __restrict__ log_dt, const float* __restrict__ A_real_log,
    const float* __restrict__ A_imag, const float* __restrict__ C_re,
    const float* __restrict__ C_im, f16* __restrict__ kh)
{
  __shared__ float dre[Ns], dimg[Ns], cdr[Ns], cdi[Ns];
  const int h = blockIdx.x;
  const int t = threadIdx.x;
  if (t < Ns) {
    float dt  = __expf(log_dt[h]);
    float aRe = -__expf(A_real_log[h * Ns + t]);
    float aIm = A_imag[h * Ns + t];
    float dr = aRe * dt, di = aIm * dt;
    float er = __expf(dr);
    float eR = er * __cosf(di) - 1.0f;   // Re(exp(dtA) - 1)
    float eI = er * __sinf(di);          // Im(exp(dtA) - 1)
    float cr = C_re[h * Ns + t], ci = C_im[h * Ns + t];
    float nr = cr * eR - ci * eI;
    float ni = cr * eI + ci * eR;
    float iden = frcp(aRe * aRe + aIm * aIm);
    dre[t]  = dr;
    dimg[t] = di;
    cdr[t] = (nr * aRe + ni * aIm) * iden;   // C_disc = C*(exp(dtA)-1)/A
    cdi[t] = (ni * aRe - nr * aIm) * iden;
  }
  __syncthreads();
  const float l = (float)t;
  float acc = 0.0f;
#pragma unroll 8
  for (int n = 0; n < Ns; ++n) {
    float mag = __expf(dre[n] * l);
    float ang = dimg[n] * l;
    acc += cdr[n] * (mag * __cosf(ang)) - cdi[n] * (mag * __sinf(ang));
  }
  kh[h * Ls + t] = (f16)(2.0f * acc);
}

// ---------------------------------------------------------------------------
// Kernel 2: repack W_out (2H x H, f32) into WMMA A-fragment order (f16):
//   WF[((gt*32 + kk)*32 + lane)*16 + e] = W[gt*16 + m][kk*32 + ka]
// where m=lane&15, half=lane>>4, ka = half*16 + e.
// Kernel 4 then loads A operands as single aligned 32B reads, no LDS needed.
// ---------------------------------------------------------------------------
__global__ __launch_bounds__(256) void w_fragize(const float* __restrict__ W,
                                                 f16* __restrict__ WF)
{
  int idx  = blockIdx.x * 256 + threadIdx.x;  // 0 .. 2*Hd*Hd-1
  int e    = idx & 15;
  int lane = (idx >> 4) & 31;
  int kk   = (idx >> 9) & 31;
  int gt   = idx >> 14;                       // g-tile 0..127
  int m = lane & 15, half = lane >> 4;
  int ka = half * 16 + e;
  WF[idx] = (f16)W[(size_t)(gt * 16 + m) * Hd + kk * 32 + ka];
}

// ---------------------------------------------------------------------------
// Kernel 3: causal conv as Toeplitz WMMA + D*x skip + GELU.
// Block = (h, 16-batch tile). Only 16 distinct Toeplitz tiles exist
// (they depend only on delta = j - 2*kk), so all are prebuilt in LDS in
// B-fragment layout; the X tile is prebuilt in A-fragment layout. The inner
// loop is then pure ds_load_b128 + v_wmma with zero branches.
// Output is written in B-fragment-friendly global layout for kernel 4:
//   ygF[((b*32 + h/32)*L + l)*32 + (h%32)]
// ---------------------------------------------------------------------------
__global__ __launch_bounds__(256) void s4d_conv_gelu(
    const float* __restrict__ x, const f16* __restrict__ kh,
    const float* __restrict__ D, f16* __restrict__ ygF)
{
  __shared__ __align__(32) f16 kpad[2 * Ls];      // [0..255]=0, [256+d]=k[d]
  __shared__ __align__(32) f16 Tt[16 * 32 * 16];  // 16 Toeplitz B-fragment tiles
  __shared__ __align__(32) f16 XsF[8 * 32 * 16];  // X A-fragments per K-step
  const int h   = blockIdx.x;
  const int b0  = blockIdx.y * 16;
  const int tid = threadIdx.x;

  kpad[tid]      = (f16)0.0f;
  kpad[Ls + tid] = kh[h * Ls + tid];
  // stage X tile directly in A-fragment order (global reads, no LDS dep)
#pragma unroll
  for (int q = 0; q < 16; ++q) {
    int idx  = tid + q * 256;                     // 0..4095
    int e    = idx & 15;
    int lane = (idx >> 4) & 31;
    int kk   = idx >> 9;                          // 0..7
    int m = lane & 15, half = lane >> 4;
    int t = kk * 32 + ((e < 8) ? e : e + 8) + half * 8;
    XsF[idx] = (f16)x[(size_t)(b0 + m) * Hd * Ls + (size_t)h * Ls + t];
  }
  __syncthreads();
  // build all 16 Toeplitz tiles in B-fragment order (branch-free via kpad)
#pragma unroll
  for (int q = 0; q < 32; ++q) {
    int idx  = tid + q * 256;                     // 0..8191
    int e    = idx & 15;
    int lane = (idx >> 4) & 31;
    int dlt  = idx >> 9;                          // delta 0..15
    int n = lane & 15, half = lane >> 4;
    int tt = ((e < 8) ? e : e + 8) + half * 8;    // B-fragment K index
    Tt[idx] = kpad[Ls + 16 * dlt + n - tt];       // k[16*dlt + n - tt] or 0
  }
  __syncthreads();

  const int wave = tid >> 5;
  const int lane = tid & 31;
  const float Dh = D[h];
  const int kkb  = h >> 5;                        // K-block for ygF layout
  const int r32  = h & 31;                        // row within K-block
  const int nn   = lane & 15;
  const int half = lane >> 4;
  const int mb   = b0 + (half ? 8 : 0);           // C/D: VGPR r <-> M=r / 8+r

  const int jlist[2] = { wave, 15 - wave };       // load-balanced: 9 WMMA/wave
#pragma unroll
  for (int jj = 0; jj < 2; ++jj) {
    const int j  = jlist[jj];
    const int l0 = j * 16;
    v8f c = {};
    const int kkmax = j >> 1;                     // causal triangular bound
    for (int kk = 0; kk <= kkmax; ++kk) {
      v16h a  = *(const v16h*)&XsF[(kk * 32 + lane) * 16];
      v16h bm = *(const v16h*)&Tt[((j - 2 * kk) * 32 + lane) * 16];
      c = __builtin_amdgcn_wmma_f32_16x16x32_f16(false, a, false, bm,
                                                 (short)0, c, false, false);
    }
    // epilogue: skip + GELU; single base pointer, r-strides fold to IOFFSET
    const int lidx = l0 + nn;
    const float* xp = x + (size_t)mb * Hd * Ls + (size_t)h * Ls + lidx;
    f16* yp = ygF + (((size_t)mb * 32 + kkb) * Ls + lidx) * 32 + r32;
#pragma unroll
    for (int r = 0; r < 8; ++r) {
      float v = gelu_fast(c[r] + Dh * xp[(size_t)r * Hd * Ls]);
      yp[(size_t)r * 32 * Ls * 32] = (f16)v;      // 512KB stride, fits IOFFSET
    }
  }
}

// ---------------------------------------------------------------------------
// Kernel 4: output projection GEMM (2H x H)@(H x L) per batch + fused GLU.
// Both operands are pre-fragmented in global memory, so the K-loop is pure
// global_load_b128 + v_wmma: no LDS, no barriers. W fragments (4MB) are
// reused 256x and stay L2-resident; ygF streams through once. The gate tile
// (row block 64+gi) is addressed off the same base via a 2MB constant offset.
// ---------------------------------------------------------------------------
__global__ __launch_bounds__(256) void out_gemm_glu(
    const f16* __restrict__ ygF, const f16* __restrict__ WF,
    const float* __restrict__ b_out, float* __restrict__ out)
{
  const int b   = blockIdx.z;
  const int gi  = blockIdx.y;                     // value tile 0..63
  const int lB  = blockIdx.x * 128;
  const int tid = threadIdx.x;
  const int wave = tid >> 5, lane = tid & 31;
  const int n = lane & 15, half = lane >> 4;
  const int lw = lB + wave * 16;                  // this wave's l-tile base

  const f16* wmat = WF  + ((size_t)gi * 32 * 32 + lane) * 16;
  const f16* yb   = ygF + (((size_t)b * 32) * Ls + lw + n) * 32
                        + (size_t)half * 16;

  v8f ca = {}, cb = {};
#pragma unroll 4
  for (int kk = 0; kk < 32; ++kk) {
    v16h aa = *(const v16h*)(wmat + (size_t)kk * 512);
    v16h ab = *(const v16h*)(wmat + (size_t)kk * 512 + (size_t)64 * 32 * 512);
    v16h bf = *(const v16h*)(yb + (size_t)kk * 8192);
    if (kk < 31)  // pull next K-step of the streamed operand into cache
      __builtin_prefetch((const void*)(yb + (size_t)(kk + 1) * 8192), 0, 1);
    ca = __builtin_amdgcn_wmma_f32_16x16x32_f16(false, aa, false, bf,
                                                (short)0, ca, false, false);
    cb = __builtin_amdgcn_wmma_f32_16x16x32_f16(false, ab, false, bf,
                                                (short)0, cb, false, false);
  }
  // epilogue: bias + GLU, write f32 output (B, H, L); strides fold to IOFFSET
  const float* bo = b_out + gi * 16 + (half ? 8 : 0);
  float* outp = out + ((size_t)b * Hd + gi * 16 + (half ? 8 : 0)) * Ls + lw + n;
#pragma unroll
  for (int r = 0; r < 8; ++r) {
    float va = ca[r] + bo[r];
    float vb = cb[r] + bo[r + Hd];
    outp[(size_t)r * Ls] = va * sigmoid_fast(vb);
  }
}

// ---------------------------------------------------------------------------
// Launcher
// ---------------------------------------------------------------------------
extern "C" void kernel_launch(void* const* d_in, const int* in_sizes, int n_in,
                              void* d_out, int out_size, void* d_ws, size_t ws_size,
                              hipStream_t stream) {
  (void)in_sizes; (void)n_in; (void)out_size; (void)ws_size;
  const float* x          = (const float*)d_in[0];
  const float* log_dt     = (const float*)d_in[1];
  const float* A_real_log = (const float*)d_in[2];
  const float* A_imag     = (const float*)d_in[3];
  const float* C_re       = (const float*)d_in[4];
  const float* C_im       = (const float*)d_in[5];
  const float* D          = (const float*)d_in[6];
  const float* W_out      = (const float*)d_in[7];
  const float* b_out      = (const float*)d_in[8];
  float* out = (float*)d_out;

  // workspace layout
  char* ws = (char*)d_ws;
  f16* kh  = (f16*)ws;                       // H*L f16             = 0.5 MB
  f16* WF  = (f16*)(ws + (1u << 20));        // 2H*H f16 fragments  = 4 MB
  f16* ygF = (f16*)(ws + (8u << 20));        // B*H*L f16 fragments = 64 MB

  s4d_kernel_build<<<Hd, 256, 0, stream>>>(log_dt, A_real_log, A_imag,
                                           C_re, C_im, kh);
  w_fragize<<<(2 * Hd * Hd) / 256, 256, 0, stream>>>(W_out, WF);
  dim3 gconv(Hd, Bsz / 16);
  s4d_conv_gelu<<<gconv, 256, 0, stream>>>(x, kh, D, ygF);
  dim3 ggemm(Ls / 128, Hd / 16, Bsz);
  out_gemm_glu<<<ggemm, 256, 0, stream>>>(ygF, WF, b_out, out);
}
